// EncoderBlock_87840671137988
// MI455X (gfx1250) — matrix operand order repaired and checked by reading, not compiled, not run on hardware
//
#include <hip/hip_runtime.h>
#include <hip/hip_bf16.h>

typedef __attribute__((ext_vector_type(16))) _Float16 v16h;
typedef __attribute__((ext_vector_type(8)))  float    v8f;

#define BATCH   4
#define C_IN    64
#define C_OUT   128
#define Hdim    96
#define Wdim    96
#define L_SEQ   (Hdim*Wdim)           // 9216
#define D_INNER 128
#define D_STATE 16
#define D_CONV  4
#define M_ROWS  (BATCH*L_SEQ)         // 36864
#define N_XZ    (2*D_INNER)           // 256
#define N_PROJ  (D_INNER + 2*D_STATE) // 160
#define K_DOWN  (C_IN*9)              // 576 = 18*32
#define OHW     48
#define M_DOWN  (BATCH*OHW*OHW)       // 9216

#if defined(__gfx1250__) && __has_builtin(__builtin_amdgcn_tensor_load_to_lds) && __has_builtin(__builtin_amdgcn_s_wait_tensorcnt)
#define USE_TDM 1
#else
#define USE_TDM 0
#endif

#if USE_TDM
// 2D f16 tile load via Tensor Data Mover: rows x width elements, source row
// stride srcStride (elements), LDS rows padded by 4 DWORDs (width must be 128:
// pad_interval code 5 = 64 DW = 256B row, pad_amount code 3 = 4 DW -> LDS row
// stride 136 halves). Waits TENSORcnt==0 before returning.
__device__ __forceinline__ void tdm_load_tile_f16(unsigned lds_addr,
                                                  const _Float16* gsrc,
                                                  int rows, int width,
                                                  int srcStride) {
  typedef unsigned int v4u __attribute__((ext_vector_type(4)));
  typedef int v4i __attribute__((ext_vector_type(4)));
  typedef int v8i __attribute__((ext_vector_type(8)));
  unsigned long long ga = (unsigned long long)gsrc;
  v4u g0;
  g0[0] = 1u;                                   // count=1 valid descriptor
  g0[1] = lds_addr;                             // lds_addr[31:0]
  g0[2] = (unsigned)ga;                         // global_addr[31:0]
  g0[3] = ((unsigned)(ga >> 32) & 0x1FFFFFFu) | (2u << 30); // addr[56:32] | type=2
  v8i g1;
  g1[0] = (int)((1u << 16) |                    // data_size = 2 bytes
                (1u << 20) |                    // pad_enable
                (5u << 22) |                    // pad_interval: 64 DWORDs
                (3u << 25));                    // pad_amount: 4 DWORDs
  g1[1] = (int)((unsigned)width << 16);         // tensor_dim0[15:0]
  g1[2] = (int)(((unsigned)M_ROWS & 0xFFFFu) << 16); // tensor_dim1[15:0]
  g1[3] = (int)((unsigned)width << 16);         // tile_dim0
  g1[4] = rows;                                 // tile_dim1 (tile_dim2=0)
  g1[5] = srcStride;                            // tensor_dim0_stride[31:0]
  g1[6] = 0;
  g1[7] = 0;
  v4i z4 = {0, 0, 0, 0};
#if __clang_major__ >= 23
  v8i z8 = {0, 0, 0, 0, 0, 0, 0, 0};
  __builtin_amdgcn_tensor_load_to_lds(g0, g1, z4, z4, z8, 0);
#else
  __builtin_amdgcn_tensor_load_to_lds(g0, g1, z4, z4, 0);
#endif
  __builtin_amdgcn_s_wait_tensorcnt((short)0);
}
#endif

// ---------------- prep kernels ----------------
__global__ void k_prep1(const float* __restrict__ ipw, const float* __restrict__ opw,
                        const float* __restrict__ Alog, const float* __restrict__ dw,
                        _Float16* __restrict__ ipwh, _Float16* __restrict__ opwh,
                        float* __restrict__ An, _Float16* __restrict__ dwh) {
  int t = blockIdx.x * 256 + threadIdx.x;
  if (t < N_XZ * C_IN)        ipwh[t] = (_Float16)ipw[t];   // [256][64]
  if (t < C_IN * D_INNER)     opwh[t] = (_Float16)opw[t];   // [64][128]
  if (t < D_INNER * D_STATE)  An[t]   = -__expf(Alog[t]);   // [128][16]
  if (t < C_OUT * K_DOWN)     dwh[t]  = (_Float16)dw[t];    // [128][576]
}

// Wc[160][128] f16: rows 0..127 = dt_proj_w @ x_proj_w[0:4,:], rows 128..159 = x_proj_w[4:36,:]
__global__ void k_prep2(const float* __restrict__ xpw, const float* __restrict__ dtw,
                        _Float16* __restrict__ Wc) {
  int t = blockIdx.x * 256 + threadIdx.x;
  if (t >= N_PROJ * D_INNER) return;
  int row = t >> 7, j = t & 127;
  float v;
  if (row < D_INNER) {
    v = 0.f;
#pragma unroll
    for (int q = 0; q < 4; q++) v += dtw[row * 4 + q] * xpw[q * D_INNER + j];
  } else {
    v = xpw[(4 + row - D_INNER) * D_INNER + j];
  }
  Wc[t] = (_Float16)v;
}

// ---------------- in_proj GEMM: xz[M,256] = seq[M,64] @ W^T + b ----------------
__global__ __launch_bounds__(256) void k_inproj(const float* __restrict__ x,
                                                const _Float16* __restrict__ Wh,
                                                const float* __restrict__ bias,
                                                float* __restrict__ xz) {
  __shared__ _Float16 As[16][72];
  const int m0 = blockIdx.x * 16;
  const int b  = m0 / L_SEQ;
  const int l0 = m0 % L_SEQ;
  const int t  = threadIdx.x;
#pragma unroll
  for (int i = 0; i < 4; i++) {
    int idx = t + i * 256;
    int k = idx >> 4, m = idx & 15;
    As[m][k] = (_Float16)x[(b * C_IN + k) * L_SEQ + l0 + m];
  }
  __syncthreads();
  const int lane = t & 31, wave = t >> 5;
  const int lm = lane & 15;
  const bool hi = lane >= 16;
  const int kb = hi ? 8 : 0, koff = hi ? 16 : 0;
  v16h a0, a1;
#pragma unroll
  for (int r = 0; r < 8; r++) {
    int k = (r < 4) ? (kb + 2 * r) : (16 + kb + 2 * (r - 4));
    a0[2*r] = As[lm][k];      a0[2*r+1] = As[lm][k+1];
    a1[2*r] = As[lm][32+k];   a1[2*r+1] = As[lm][32+k+1];
  }
  for (int nt = wave; nt < 16; nt += 8) {
    int n0 = nt * 16, nn = n0 + lm;
    v16h b0, b1;
#pragma unroll
    for (int r = 0; r < 8; r++) {
      int k = (r < 4) ? (kb + 2 * r) : (16 + kb + 2 * (r - 4));
      b0[2*r] = Wh[nn*64 + k];      b0[2*r+1] = Wh[nn*64 + k + 1];
      b1[2*r] = Wh[nn*64 + 32 + k]; b1[2*r+1] = Wh[nn*64 + 32 + k + 1];
    }
    v8f c = {};
    c = __builtin_amdgcn_wmma_f32_16x16x32_f16(false, a0, false, b0, (short)0, c, false, false);
    c = __builtin_amdgcn_wmma_f32_16x16x32_f16(false, a1, false, b1, (short)0, c, false, false);
#pragma unroll
    for (int r = 0; r < 8; r++) {
      int m = m0 + r + (hi ? 8 : 0);
      int n = n0 + lm;
      xz[(size_t)m * N_XZ + n] = c[r] + bias[n];
    }
  }
}

// ---------------- depthwise causal conv1d + SiLU (f32 + f16 outputs) ----------------
__global__ __launch_bounds__(256) void k_conv(const float* __restrict__ xz,
                                              const float* __restrict__ cw,
                                              const float* __restrict__ cb,
                                              float* __restrict__ xcs,
                                              _Float16* __restrict__ xcsh) {
  int idx = blockIdx.x * 256 + threadIdx.x;
  if (idx >= M_ROWS * D_INNER) return;
  int d = idx & 127;
  int bl = idx >> 7;
  int b = bl / L_SEQ, l = bl % L_SEQ;
  float acc = cb[d];
#pragma unroll
  for (int j = 0; j < D_CONV; j++) {
    int ls = l - (D_CONV - 1) + j;
    if (ls >= 0) acc += cw[d * D_CONV + j] * xz[(size_t)(b * L_SEQ + ls) * N_XZ + d];
  }
  float s = acc / (1.f + __expf(-acc));
  xcs[idx]  = s;
  xcsh[idx] = (_Float16)s;
}

// ---------------- fused x_proj+dt_proj GEMM: proj[M,160] = xcs[M,128] @ Wc^T ----------------
__global__ __launch_bounds__(256) void k_xproj(const _Float16* __restrict__ xcsh,
                                               const _Float16* __restrict__ Wc,
                                               const float* __restrict__ dtb,
                                               float* __restrict__ proj) {
  __shared__ __align__(16) _Float16 As[16][136];
  const int m0 = blockIdx.x * 16;
  const int t = threadIdx.x;
#if USE_TDM
  if (t < 32) {   // wave 0 issues the TDM load of the 16x128 f16 tile
    tdm_load_tile_f16((unsigned)(unsigned long long)(void*)&As[0][0],
                      xcsh + (size_t)m0 * D_INNER, 16, D_INNER, D_INNER);
  }
  __syncthreads();
#else
  {
    const unsigned* xu = (const unsigned*)(xcsh + (size_t)m0 * D_INNER);
    unsigned* au = (unsigned*)&As[0][0];
#pragma unroll
    for (int i = 0; i < 4; i++) {
      int idx = t + i * 256;            // 1024 dwords
      int m = idx >> 6, kk = idx & 63;
      au[m * 68 + kk] = xu[m * 64 + kk];
    }
  }
  __syncthreads();
#endif
  const int lane = t & 31, wave = t >> 5;
  const int lm = lane & 15;
  const bool hi = lane >= 16;
  const int kb = hi ? 8 : 0, koff = hi ? 16 : 0;
  v16h af[4];
#pragma unroll
  for (int ks = 0; ks < 4; ks++)
#pragma unroll
    for (int r = 0; r < 8; r++) {
      int k = ks * 32 + ((r < 4) ? (kb + 2 * r) : (16 + kb + 2 * (r - 4)));
      af[ks][2*r] = As[lm][k]; af[ks][2*r+1] = As[lm][k+1];
    }
  for (int nt = wave; nt < 10; nt += 8) {
    int n0 = nt * 16, nn = n0 + lm;
    v8f c = {};
#pragma unroll
    for (int ks = 0; ks < 4; ks++) {
      v16h bf;
#pragma unroll
      for (int r = 0; r < 8; r++) {
        int k = ks * 32 + ((r < 4) ? (kb + 2 * r) : (16 + kb + 2 * (r - 4)));
        bf[2*r] = Wc[nn*128 + k]; bf[2*r+1] = Wc[nn*128 + k + 1];
      }
      c = __builtin_amdgcn_wmma_f32_16x16x32_f16(false, af[ks], false, bf, (short)0, c, false, false);
    }
#pragma unroll
    for (int r = 0; r < 8; r++) {
      int m = m0 + r + (hi ? 8 : 0);
      int n = n0 + lm;
      float v = c[r];
      if (n < D_INNER) {                  // delta columns: softplus(v + dt_bias)
        v += dtb[n];
        v = (v > 20.f) ? v : __logf(1.f + __expf(v));
      }
      proj[(size_t)m * N_PROJ + n] = v;
    }
  }
}

// ---------------- selective scan (y emitted as f16 for the out_proj GEMM) ----------------
__global__ __launch_bounds__(256) void k_scan(const float* __restrict__ proj,
                                              const float* __restrict__ xcs,
                                              const float* __restrict__ xz,
                                              const float* __restrict__ An,
                                              const float* __restrict__ Dp,
                                              _Float16* __restrict__ yh) {
  const int t = threadIdx.x;
  const int lane = t & 31, wave = t >> 5;
  const int gw = blockIdx.x * 8 + wave;           // 0..255
  const int pair = gw * 2 + (lane >= 16 ? 1 : 0); // 0..511
  const int b = pair >> 7, d = pair & 127;
  const int n = lane & 15;
  const float a  = An[d * D_STATE + n];
  const float Dd = Dp[d];
  float h = 0.f;
  for (int l = 0; l < L_SEQ; l++) {
    const size_t row = (size_t)b * L_SEQ + l;
    const float* pr = proj + row * N_PROJ;
    __builtin_prefetch((const void*)(pr + 16 * N_PROJ), 0, 1);
    float delta = pr[d];
    float Bn = pr[D_INNER + n];
    float Cn = pr[D_INNER + D_STATE + n];
    float u  = xcs[row * D_INNER + d];
    h = __expf(delta * a) * h + delta * Bn * u;
    float yc = h * Cn;
    yc += __shfl_xor(yc, 1, 32);
    yc += __shfl_xor(yc, 2, 32);
    yc += __shfl_xor(yc, 4, 32);
    yc += __shfl_xor(yc, 8, 32);
    if (n == 0) {
      float res = xz[row * N_XZ + D_INNER + d];
      float sil = res / (1.f + __expf(-res));
      yh[row * D_INNER + d] = (_Float16)((yc + u * Dd) * sil);
    }
  }
}

// ---------------- out_proj GEMM + LayerNorm(C) + SiLU fused ----------------
__global__ __launch_bounds__(256) void k_outln(const _Float16* __restrict__ yh,
                                               const _Float16* __restrict__ Wo,
                                               const float* __restrict__ ob,
                                               const float* __restrict__ g,
                                               const float* __restrict__ be,
                                               float* __restrict__ r_out) {
  __shared__ __align__(16) _Float16 As[8][16][136];
  const int t = threadIdx.x;
  const int lane = t & 31, wave = t >> 5;
  const int mt = blockIdx.x * 8 + wave;     // 0..2303
  const int m0 = mt * 16;
#if USE_TDM
  // each wave DMAs its own 16x128 f16 tile into its padded LDS slot
  tdm_load_tile_f16((unsigned)(unsigned long long)(void*)&As[wave][0][0],
                    yh + (size_t)m0 * D_INNER, 16, D_INNER, D_INNER);
  __syncthreads();
#else
  {
    const unsigned* yu = (const unsigned*)(yh + (size_t)m0 * D_INNER);
    unsigned* au = (unsigned*)&As[wave][0][0];
#pragma unroll
    for (int i = 0; i < 32; i++) {
      int idx = lane + i * 32;            // 1024 dwords per wave tile
      int m = idx >> 6, kk = idx & 63;
      au[m * 68 + kk] = yu[m * 64 + kk];
    }
  }
  __syncthreads();
#endif
  const int lm = lane & 15;
  const bool hi = lane >= 16;
  const int kb = hi ? 8 : 0, koff = hi ? 16 : 0;
  v8f zero = {0,0,0,0,0,0,0,0};
  v8f acc[4] = {zero, zero, zero, zero};
  for (int ks = 0; ks < 4; ks++) {
    v16h af;
#pragma unroll
    for (int r = 0; r < 8; r++) {
      int k = ks * 32 + ((r < 4) ? (kb + 2 * r) : (16 + kb + 2 * (r - 4)));
      af[2*r] = As[wave][lm][k]; af[2*r+1] = As[wave][lm][k+1];
    }
#pragma unroll
    for (int nt = 0; nt < 4; nt++) {
      int nn = nt * 16 + lm;
      v16h bf;
#pragma unroll
      for (int r = 0; r < 8; r++) {
        int k = ks * 32 + ((r < 4) ? (kb + 2 * r) : (16 + kb + 2 * (r - 4)));
        bf[2*r] = Wo[nn*128 + k]; bf[2*r+1] = Wo[nn*128 + k + 1];
      }
      acc[nt] = __builtin_amdgcn_wmma_f32_16x16x32_f16(false, af, false, bf, (short)0, acc[nt], false, false);
    }
  }
#pragma unroll
  for (int nt = 0; nt < 4; nt++)
#pragma unroll
    for (int r = 0; r < 8; r++) acc[nt][r] += ob[nt * 16 + lm];
#pragma unroll
  for (int r = 0; r < 8; r++) {
    float s  = acc[0][r] + acc[1][r] + acc[2][r] + acc[3][r];
    float s2 = acc[0][r]*acc[0][r] + acc[1][r]*acc[1][r] + acc[2][r]*acc[2][r] + acc[3][r]*acc[3][r];
#pragma unroll
    for (int msk = 1; msk < 16; msk <<= 1) {
      s  += __shfl_xor(s,  msk, 32);
      s2 += __shfl_xor(s2, msk, 32);
    }
    float mu  = s * (1.f / 64.f);
    float var = s2 * (1.f / 64.f) - mu * mu;
    float rs  = rsqrtf(var + 1e-5f);
    int m = m0 + r + (hi ? 8 : 0);
    int bb = m / L_SEQ, l = m % L_SEQ;
#pragma unroll
    for (int nt = 0; nt < 4; nt++) {
      int n = nt * 16 + lm;
      float v = (acc[nt][r] - mu) * rs * g[n] + be[n];
      v = v / (1.f + __expf(-v));
      r_out[(size_t)(bb * C_IN + n) * L_SEQ + l] = v;
    }
  }
}

// ---------------- downsample conv 3x3 s2 p1 as WMMA implicit GEMM ----------------
// out[M_DOWN,128] = patches[M_DOWN, 576] @ Wd[128,576]^T, K = 576 = 18*32
__global__ __launch_bounds__(256) void k_down(const float* __restrict__ r,
                                              const _Float16* __restrict__ wdh,
                                              const float* __restrict__ bias,
                                              float* __restrict__ out) {
  const int t = threadIdx.x, lane = t & 31, wave = t >> 5;
  const int mt = blockIdx.x * 8 + wave;     // 0..575
  const int m0 = mt * 16;
  const int lm = lane & 15;
  const bool hi = lane >= 16;
  const int kb = hi ? 8 : 0, koff = hi ? 16 : 0;
  // this lane's A-row (for gathers)
  const int mA = m0 + lm;
  const int bA  = mA / (OHW * OHW);
  const int ohA = (mA % (OHW * OHW)) / OHW;
  const int owA = mA % OHW;
  const int ih0 = ohA * 2 - 1, iw0 = owA * 2 - 1;
  const float* rb = r + (size_t)bA * C_IN * L_SEQ;
  v8f zero = {0,0,0,0,0,0,0,0};
  v8f acc[8] = {zero, zero, zero, zero, zero, zero, zero, zero};
  for (int ks = 0; ks < 18; ks++) {
    v16h af;
#pragma unroll
    for (int rr = 0; rr < 8; rr++) {
#pragma unroll
      for (int e = 0; e < 2; e++) {
        int k = ks * 32 + ((rr < 4) ? (kb + 2*rr) : (16 + kb + 2*(rr-4))) + e;
        int ci = k / 9, rem = k % 9;
        int kh = rem / 3, kw = rem % 3;
        int ih = ih0 + kh, iw = iw0 + kw;
        float v = 0.f;
        if (ih >= 0 && ih < Hdim && iw >= 0 && iw < Wdim)
          v = rb[ci * L_SEQ + ih * Wdim + iw];
        af[2*rr + e] = (_Float16)v;
      }
    }
#pragma unroll
    for (int nt = 0; nt < 8; nt++) {
      int nn = nt * 16 + lm;
      v16h bf;
#pragma unroll
      for (int rr = 0; rr < 8; rr++) {
        int k = ks * 32 + ((rr < 4) ? (kb + 2*rr) : (16 + kb + 2*(rr-4)));
        bf[2*rr]   = wdh[nn * K_DOWN + k];
        bf[2*rr+1] = wdh[nn * K_DOWN + k + 1];
      }
      acc[nt] = __builtin_amdgcn_wmma_f32_16x16x32_f16(false, af, false, bf, (short)0, acc[nt], false, false);
    }
  }
#pragma unroll
  for (int rr = 0; rr < 8; rr++) {
    int m = m0 + rr + (hi ? 8 : 0);
    int b = m / (OHW * OHW), oh = (m % (OHW * OHW)) / OHW, ow = m % OHW;
#pragma unroll
    for (int nt = 0; nt < 8; nt++) {
      int n = nt * 16 + lm;
      out[(size_t)((b * C_OUT + n) * OHW + oh) * OHW + ow] = acc[nt][rr] + bias[n];
    }
  }
}

// ---------------- host launch ----------------
extern "C" void kernel_launch(void* const* d_in, const int* in_sizes, int n_in,
                              void* d_out, int out_size, void* d_ws, size_t ws_size,
                              hipStream_t stream) {
  const float* x          = (const float*)d_in[0];
  const float* in_proj_w  = (const float*)d_in[1];
  const float* in_proj_b  = (const float*)d_in[2];
  const float* conv1d_w   = (const float*)d_in[3];
  const float* conv1d_b   = (const float*)d_in[4];
  const float* x_proj_w   = (const float*)d_in[5];
  const float* dt_proj_w  = (const float*)d_in[6];
  const float* dt_proj_b  = (const float*)d_in[7];
  const float* A_log      = (const float*)d_in[8];
  const float* Dp         = (const float*)d_in[9];
  const float* out_proj_w = (const float*)d_in[10];
  const float* out_proj_b = (const float*)d_in[11];
  const float* ln_g       = (const float*)d_in[12];
  const float* ln_b       = (const float*)d_in[13];
  const float* down_w     = (const float*)d_in[14];
  const float* down_b     = (const float*)d_in[15];

  char* ws = (char*)d_ws;
  const size_t XZ_OFF   = 0;
  const size_t XCS_OFF  = XZ_OFF   + (size_t)M_ROWS * N_XZ    * 4;
  const size_t PROJ_OFF = XCS_OFF  + (size_t)M_ROWS * D_INNER * 4;
  const size_t XCSH_OFF = PROJ_OFF + (size_t)M_ROWS * N_PROJ  * 4;
  const size_t YH_OFF   = XCSH_OFF + (size_t)M_ROWS * D_INNER * 2;
  const size_t IPWH_OFF = YH_OFF   + (size_t)M_ROWS * D_INNER * 2;
  const size_t OPWH_OFF = IPWH_OFF + (size_t)N_XZ * C_IN * 2;
  const size_t WC_OFF   = OPWH_OFF + (size_t)C_IN * D_INNER * 2;
  const size_t DWH_OFF  = WC_OFF   + (size_t)N_PROJ * D_INNER * 2;
  const size_t AN_OFF   = DWH_OFF  + (size_t)C_OUT * K_DOWN * 2;

  float*    xz   = (float*)(ws + XZ_OFF);
  float*    xcs  = (float*)(ws + XCS_OFF);
  float*    proj = (float*)(ws + PROJ_OFF);
  _Float16* xcsh = (_Float16*)(ws + XCSH_OFF);
  _Float16* yh   = (_Float16*)(ws + YH_OFF);
  _Float16* ipwh = (_Float16*)(ws + IPWH_OFF);
  _Float16* opwh = (_Float16*)(ws + OPWH_OFF);
  _Float16* wc   = (_Float16*)(ws + WC_OFF);
  _Float16* dwh  = (_Float16*)(ws + DWH_OFF);
  float*    An   = (float*)(ws + AN_OFF);

  float* down_out = (float*)d_out;
  float* r_out    = down_out + (size_t)BATCH * C_OUT * OHW * OHW;

  k_prep1<<<(C_OUT * K_DOWN + 255) / 256, 256, 0, stream>>>(
      in_proj_w, out_proj_w, A_log, down_w, ipwh, opwh, An, dwh);
  k_prep2<<<(N_PROJ * D_INNER + 255) / 256, 256, 0, stream>>>(x_proj_w, dt_proj_w, wc);
  k_inproj<<<M_ROWS / 16, 256, 0, stream>>>(x, ipwh, in_proj_b, xz);
  k_conv<<<(M_ROWS * D_INNER + 255) / 256, 256, 0, stream>>>(xz, conv1d_w, conv1d_b, xcs, xcsh);
  k_xproj<<<M_ROWS / 16, 256, 0, stream>>>(xcsh, wc, dt_proj_b, proj);
  k_scan<<<32, 256, 0, stream>>>(proj, xcs, xz, An, Dp, yh);
  k_outln<<<M_ROWS / (16 * 8), 256, 0, stream>>>(yh, opwh, out_proj_b, ln_g, ln_b, r_out);
  k_down<<<M_DOWN / (16 * 8), 256, 0, stream>>>(r_out, dwh, down_b, down_out);
}
